// MultiHeadAttention_40097814675883
// MI455X (gfx1250) — matrix operand order repaired and checked by reading, not compiled, not run on hardware
//
#include <hip/hip_runtime.h>
#include <stdint.h>

typedef unsigned short u16;
typedef unsigned int   u32;
typedef __attribute__((ext_vector_type(16))) __bf16 v16bf;
typedef __attribute__((ext_vector_type(8)))  float  v8f;

// Problem constants (B=2, S=2048, D=1024, H=16, HD=64)
constexpr int kB = 2, kS = 2048, kD = 1024, kH = 16, kHD = 64;

union FragAB {
    v16bf v;
    uint4 q[2];
};

__device__ __forceinline__ u16 f2bf(float f) {
    u32 u = __builtin_bit_cast(u32, f);
    u32 r = (u + 0x7FFFu + ((u >> 16) & 1u)) >> 16;  // round-to-nearest-even
    return (u16)r;
}
__device__ __forceinline__ u32 pack2(float x, float y) {
    return (u32)f2bf(x) | ((u32)f2bf(y) << 16);
}

// ---------------------------------------------------------------------------
// Kernel 1: fused projection  out = bf16(X @ W + bias), scattered to
//   mode 0: [B,H,S,HD]   (Q, K)
//   mode 1: [B,H,HD,S]   (V transposed, so PV B-fragments are contiguous)
// Block tile 64x64, BK=32 (== WMMA K). 8 waves, each wave = 16x32 C strip.
// ---------------------------------------------------------------------------
__global__ __launch_bounds__(256) void qkv_proj_kernel(
    const float* __restrict__ X, const float* __restrict__ W,
    const float* __restrict__ bias, u16* __restrict__ out, int transposed)
{
    __shared__ u16 Xs[64][40];   // [m][k] bf16, row stride 80B (16B aligned)
    __shared__ u16 Ws[64][40];   // [n][k] bf16 (transposed W tile)

    const int tid  = threadIdx.x;
    const int lane = tid & 31;
    const int wid  = tid >> 5;
    const int l16  = lane & 15;
    const bool hi  = lane >= 16;

    const int m0 = blockIdx.y * 64;
    const int n0 = blockIdx.x * 64;
    const int rowBase = (wid >> 1) * 16;   // 4 waves along M
    const int colBase = (wid & 1) * 32;    // 2 waves along N

    const int xrow = tid >> 2, xcol = (tid & 3) * 8;   // 64x32 f32, 8/thread
    const int wrow = tid >> 3, wcol = (tid & 7) * 8;   // 32x64 f32, 8/thread

    v8f acc[2] = {};

    for (int k0 = 0; k0 < kD; k0 += 32) {
        const float* px = X + (m0 + xrow) * kD + k0 + xcol;
        const float* pw = W + (k0 + wrow) * kD + n0 + wcol;
        // Warm L2/WGP$ one k-tile ahead (global_prefetch_b8)
        if (k0 + 32 < kD) {
            __builtin_prefetch(px + 32, 0, 1);
            __builtin_prefetch(pw + 32 * kD, 0, 1);
        }
        // Stage X tile (fp32 -> bf16, row-major)
        {
            float4 a0 = *(const float4*)px;
            float4 a1 = *(const float4*)(px + 4);
            uint4 v;
            v.x = pack2(a0.x, a0.y); v.y = pack2(a0.z, a0.w);
            v.z = pack2(a1.x, a1.y); v.w = pack2(a1.z, a1.w);
            *(uint4*)&Xs[xrow][xcol] = v;
        }
        // Stage W tile transposed: Ws[n][k]
        {
            float4 a0 = *(const float4*)pw;
            float4 a1 = *(const float4*)(pw + 4);
            Ws[wcol + 0][wrow] = f2bf(a0.x);
            Ws[wcol + 1][wrow] = f2bf(a0.y);
            Ws[wcol + 2][wrow] = f2bf(a0.z);
            Ws[wcol + 3][wrow] = f2bf(a0.w);
            Ws[wcol + 4][wrow] = f2bf(a1.x);
            Ws[wcol + 5][wrow] = f2bf(a1.y);
            Ws[wcol + 6][wrow] = f2bf(a1.z);
            Ws[wcol + 7][wrow] = f2bf(a1.w);
        }
        __syncthreads();

        // A fragment: lane<16 -> row=lane, K 0-7 & 16-23; lane>=16 -> row=lane-16, K 8-15 & 24-31
        FragAB a;
        {
            const u16* xr = &Xs[rowBase + l16][hi ? 8 : 0];
            a.q[0] = *(const uint4*)xr;
            a.q[1] = *(const uint4*)(xr + 16);
        }
        #pragma unroll
        for (int j = 0; j < 2; ++j) {
            // B fragment: lane -> col n, K 0-7 & 16-23 (lo half) / 8-15 & 24-31 (hi half)
            FragAB bfr;
            const u16* wr = &Ws[colBase + j * 16 + l16][hi ? 8 : 0];
            bfr.q[0] = *(const uint4*)wr;
            bfr.q[1] = *(const uint4*)(wr + 16);
            acc[j] = __builtin_amdgcn_wmma_f32_16x16x32_bf16(
                false, a.v, false, bfr.v, (short)0, acc[j], false, false);
        }
        __syncthreads();
    }

    // Epilogue: bias add + bf16 + head-split scatter
    #pragma unroll
    for (int j = 0; j < 2; ++j) {
        const int c  = n0 + colBase + j * 16 + l16;
        const float bv = bias[c];
        const int h = c >> 6, d = c & 63;
        #pragma unroll
        for (int r = 0; r < 8; ++r) {
            const int rg = m0 + rowBase + r + (hi ? 8 : 0);
            const int bi = rg >> 11, s = rg & 2047;
            const float val = acc[j][r] + bv;
            const int idx = transposed
                ? (((bi * kH + h) * kHD + d) * kS + s)
                : (((bi * kH + h) * kS + s) * kHD + d);
            out[idx] = f2bf(val);
        }
    }
}

// ---------------------------------------------------------------------------
// Kernel 2: flash-style attention. Block = (b,h, 128 queries); 8 waves,
// each wave owns 16 queries. Keys streamed in blocks of 32 with
// double-buffered GLOBAL_LOAD_ASYNC_TO_LDS_B128 (ASYNCcnt-tracked, no VGPR
// round trip), overlapping the next tile's fill with this tile's WMMAs.
// ---------------------------------------------------------------------------
__global__ __launch_bounds__(256) void attn_kernel(
    const u16* __restrict__ Q, const u16* __restrict__ K,
    const u16* __restrict__ Vt, float* __restrict__ out)
{
    __shared__ u16 Ks[2][32][72];   // [buf][key][d]  (B-frag for QK^T)
    __shared__ u16 Vs[2][64][40];   // [buf][d][key]  (B-frag for PV)
    __shared__ u16 Ps[8][16][40];   // per-wave P tile [q][key]

    const int tid  = threadIdx.x;
    const int lane = tid & 31;
    const int wid  = tid >> 5;
    const int l16  = lane & 15;
    const bool hi  = lane >= 16;

    const int bh = blockIdx.y;           // b*16 + h
    const int b  = bh >> 4, h = bh & 15;
    const int q0 = blockIdx.x * 128 + wid * 16;

    const u16* Qh = Q  + bh * (kS * kHD);
    const u16* Kh = K  + bh * (kS * kHD);
    const u16* Vh = Vt + bh * (kHD * kS);

    // Q A-fragments (K=64 split in two 32-chunks), held in registers
    FragAB qa[2];
    #pragma unroll
    for (int kh = 0; kh < 2; ++kh) {
        const u16* qr = Qh + (q0 + l16) * kHD + kh * 32 + (hi ? 8 : 0);
        qa[kh].q[0] = *(const uint4*)qr;
        qa[kh].q[1] = *(const uint4*)(qr + 16);
    }

    float m[8], l[8];
    v8f o[4] = {};
    #pragma unroll
    for (int r = 0; r < 8; ++r) { m[r] = -1e30f; l[r] = 0.f; }

    const int kkey = tid >> 3, kc = (tid & 7) * 8;   // K tile 32x64, 16B/thread
    const int vd   = tid >> 2, vp = (tid & 3) * 8;   // V^T tile 64x32, 16B/thread

    // Async memory->LDS fill of one (K,V) tile pair. GVS addressing:
    // lds_addr_vgpr, 32-bit voffset vgpr, 64-bit saddr sgpr pair.
    auto issue_tiles = [&](int k0, int buf) {
        u32 ldsK = (u32)(uintptr_t)&Ks[buf][kkey][kc];
        u32 offK = (u32)(((k0 + kkey) * kHD + kc) * 2);
        asm volatile("global_load_async_to_lds_b128 %0, %1, %2"
                     :: "v"(ldsK), "v"(offK), "s"(Kh) : "memory");
        u32 ldsV = (u32)(uintptr_t)&Vs[buf][vd][vp];
        u32 offV = (u32)((vd * kS + k0 + vp) * 2);
        asm volatile("global_load_async_to_lds_b128 %0, %1, %2"
                     :: "v"(ldsV), "v"(offV), "s"(Vh) : "memory");
    };

    constexpr int NIT = kS / 32;
    issue_tiles(0, 0);

    for (int it = 0; it < NIT; ++it) {
        const int buf = it & 1;
        // Own async loads done -> barrier -> everyone's loads visible.
        asm volatile("s_wait_asynccnt 0" ::: "memory");
        __syncthreads();
        if (it + 1 < NIT) issue_tiles((it + 1) * 32, buf ^ 1);

        // Scores: S[16x32] = Q(16x64) . K^T(64x32)
        v8f sc[2];
        #pragma unroll
        for (int j = 0; j < 2; ++j) {
            v8f s = {};
            #pragma unroll
            for (int kh = 0; kh < 2; ++kh) {
                FragAB kb;
                const u16* kr = &Ks[buf][j * 16 + l16][kh * 32 + (hi ? 8 : 0)];
                kb.q[0] = *(const uint4*)kr;
                kb.q[1] = *(const uint4*)(kr + 16);
                s = __builtin_amdgcn_wmma_f32_16x16x32_bf16(
                    false, qa[kh].v, false, kb.v, (short)0, s, false, false);
            }
            sc[j] = s;
        }

        // Online softmax. Row M=r lives in lanes 0-15 (vgpr r) and row M=8+r
        // in lanes 16-31, so width-16 xor reductions are per-row.
        float alpha[8];
        #pragma unroll
        for (int r = 0; r < 8; ++r) {
            float s0 = sc[0][r] * 0.125f;   // 1/sqrt(64)
            float s1 = sc[1][r] * 0.125f;
            float mx = fmaxf(s0, s1);
            #pragma unroll
            for (int off = 8; off >= 1; off >>= 1)
                mx = fmaxf(mx, __shfl_xor(mx, off, 16));
            const float mn = fmaxf(m[r], mx);
            const float a  = __expf(m[r] - mn);
            const float p0 = __expf(s0 - mn);
            const float p1 = __expf(s1 - mn);
            float t = p0 + p1;
            #pragma unroll
            for (int off = 8; off >= 1; off >>= 1)
                t += __shfl_xor(t, off, 16);
            l[r] = l[r] * a + t;
            m[r] = mn;
            alpha[r] = a;
            const int rowM = r + (hi ? 8 : 0);
            Ps[wid][rowM][l16]      = f2bf(p0);
            Ps[wid][rowM][16 + l16] = f2bf(p1);
        }

        // Rescale running O by alpha (per-row, uniform across the 16-lane group)
        #pragma unroll
        for (int n = 0; n < 4; ++n)
            #pragma unroll
            for (int r = 0; r < 8; ++r)
                o[n][r] *= alpha[r];

        // Re-gather P as an A fragment (LDS ops are in-order per wave)
        FragAB pa;
        {
            const u16* pr = &Ps[wid][l16][hi ? 8 : 0];
            pa.q[0] = *(const uint4*)pr;
            pa.q[1] = *(const uint4*)(pr + 16);
        }

        // O += P(16x32) . V(32x64): 4 WMMAs over the 4 HD column blocks
        #pragma unroll
        for (int n = 0; n < 4; ++n) {
            FragAB vb;
            const u16* vr = &Vs[buf][n * 16 + l16][hi ? 8 : 0];
            vb.q[0] = *(const uint4*)vr;
            vb.q[1] = *(const uint4*)(vr + 16);
            o[n] = __builtin_amdgcn_wmma_f32_16x16x32_bf16(
                false, pa.v, false, vb.v, (short)0, o[n], false, false);
        }
    }

    // Normalize and write fp32 output [B,S,D] with head merge
    #pragma unroll
    for (int n = 0; n < 4; ++n) {
        #pragma unroll
        for (int r = 0; r < 8; ++r) {
            const int rowM = r + (hi ? 8 : 0);
            const int q = q0 + rowM;
            const float val = o[n][r] / l[r];
            out[(b * kS + q) * kD + h * kHD + n * 16 + l16] = val;
        }
    }
}

// ---------------------------------------------------------------------------
extern "C" void kernel_launch(void* const* d_in, const int* in_sizes, int n_in,
                              void* d_out, int out_size, void* d_ws, size_t ws_size,
                              hipStream_t stream) {
    (void)in_sizes; (void)n_in; (void)out_size; (void)ws_size;
    const float* query = (const float*)d_in[0];
    const float* key   = (const float*)d_in[1];
    const float* value = (const float*)d_in[2];
    const float* Wq    = (const float*)d_in[3];
    const float* bq    = (const float*)d_in[4];
    const float* Wk    = (const float*)d_in[5];
    const float* bk    = (const float*)d_in[6];
    const float* Wv    = (const float*)d_in[7];
    const float* bv    = (const float*)d_in[8];

    const size_t headElems = (size_t)kB * kH * kS * kHD;  // 4,194,304
    u16* Qb = (u16*)d_ws;
    u16* Kb = Qb + headElems;
    u16* Vb = Kb + headElems;   // V stored transposed [B,H,HD,S]

    dim3 pgrid(kD / 64, (kB * kS) / 64);   // (16, 64)
    qkv_proj_kernel<<<pgrid, 256, 0, stream>>>(query, Wq, bq, Qb, 0);
    qkv_proj_kernel<<<pgrid, 256, 0, stream>>>(key,   Wk, bk, Kb, 0);
    qkv_proj_kernel<<<pgrid, 256, 0, stream>>>(value, Wv, bv, Vb, 1);

    dim3 agrid(kS / 128, kB * kH);         // (16, 32)
    attn_kernel<<<agrid, 256, 0, stream>>>(Qb, Kb, Vb, (float*)d_out);
}
